// HAN_23356032155689
// MI455X (gfx1250) — compile-verified, matrix operand order checked by reading, not run on hardware
//
#include <hip/hip_runtime.h>
#include <hip/hip_bf16.h>
#include <math.h>

typedef __attribute__((ext_vector_type(16))) __bf16 v16bf;
typedef __attribute__((ext_vector_type(8)))  __bf16 v8bf;
typedef __attribute__((ext_vector_type(8)))  float  v8f;

#define N_NODES 10000
#define N_EDGES 320000
#define DIM_IN  384
#define HEADS   8
#define HID     64
#define D1      512
#define SA_HIDN 128
#define OUT_DIM 16
#define MP      10112   // N_NODES padded to 128 (rows)
#define KP_ADJ  10048   // N_NODES padded to 64 (K)

__device__ __forceinline__ unsigned short f2bf(float x) {
  unsigned u = __float_as_uint(x);
  unsigned r = u + 0x7FFFu + ((u >> 16) & 1u);   // round-to-nearest-even
  return (unsigned short)(r >> 16);
}

__device__ __forceinline__ void atomicMaxF(float* addr, float v) {
  if (v >= 0.f) atomicMax((int*)addr, __float_as_int(v));
  else          atomicMin((unsigned int*)addr, __float_as_uint(v));
}

// one async global->LDS 128b copy (ASYNCcnt-tracked)
#define ASYNC_B128(ldsOff, gaddr)                                             \
  asm volatile("global_load_async_to_lds_b128 %0, %1, off"                    \
               :: "v"(ldsOff), "v"(gaddr) : "memory")

#define SHUF16(lo, hi) __builtin_shufflevector(lo, hi,                        \
    0,1,2,3,4,5,6,7,8,9,10,11,12,13,14,15)

// ---------------------------------------------------------------------------
// f32 [R,C] -> bf16 padded [Rp,Cp] (zero fill)
__global__ void conv_pad_bf16(const float* __restrict__ in,
                              unsigned short* __restrict__ out,
                              int R, int C, int Rp, int Cp) {
  size_t i = (size_t)blockIdx.x * blockDim.x + threadIdx.x;
  size_t tot = (size_t)Rp * Cp;
  if (i >= tot) return;
  int r = (int)(i / Cp), c = (int)(i % Cp);
  float v = (r < R && c < C) ? in[(size_t)r * C + c] : 0.f;
  out[i] = f2bf(v);
}

// f32 [R,C] -> bf16 transposed+padded [C,Rp] (zero fill along Rp)
__global__ void conv_padT_bf16(const float* __restrict__ in,
                               unsigned short* __restrict__ out,
                               int R, int C, int Rp) {
  size_t i = (size_t)blockIdx.x * blockDim.x + threadIdx.x;
  size_t tot = (size_t)C * Rp;
  if (i >= tot) return;
  int c = (int)(i / Rp), r = (int)(i % Rp);
  float v = (r < R) ? in[(size_t)r * C + c] : 0.f;
  out[i] = f2bf(v);
}

// ---------------------------------------------------------------------------
// GEMM  C[M,Nc] = A[Mp,Kp](bf16, lda) @ BT[Nc,Kp](bf16)^T  -> f32 C row-major
// Block 256 thr = 8 waves; tile 128x128, BK=64 (two WMMA K-steps per barrier),
// double-buffered LDS (64 KB) filled by global_load_async_to_lds_b128 with a
// software-pipelined, branch-free steady-state loop. Wave w: 16x128 strip ->
// 8 accumulators, 16 v_wmma_f32_16x16x32_bf16 per iteration. B fragments are
// preloaded so the 8 WMMAs issue back-to-back.
// ---------------------------------------------------------------------------
__global__ __launch_bounds__(256)
void gemm_bf16_async(const unsigned short* __restrict__ A,
                     const unsigned short* __restrict__ BT,
                     float* __restrict__ C, int M, int Nc, int Kp, int lda) {
  __shared__ __align__(16) unsigned short As[2][128 * 64];  // [row][k]
  __shared__ __align__(16) unsigned short Bs[2][128 * 64];  // [n][k]

  const int tid  = threadIdx.x;
  const int lane = tid & 31;
  const int wave = tid >> 5;
  const int m0   = blockIdx.y * 128;
  const int n0   = blockIdx.x * 128;
  const int am   = lane & 15;
  const int cl   = lane >> 4;

  v8f acc[8] = {};

  // async copy slots: tile = 1024 x 16B chunks; 4 per thread (rows r0+32i)
  const int r0 = tid >> 3;              // 0..31
  const int kc = (tid & 7) * 8;         // bf16 col within 64-wide row
  unsigned long long gaA[4], gaB[4];
  unsigned ldsA[4], ldsB[4];
  const unsigned asBase = (unsigned)(uintptr_t)(&As[0][0]);
  const unsigned bsBase = (unsigned)(uintptr_t)(&Bs[0][0]);
  #pragma unroll
  for (int i = 0; i < 4; ++i) {
    int r = r0 + 32 * i;
    gaA[i] = (unsigned long long)(uintptr_t)(A  + (size_t)(m0 + r) * lda + kc);
    gaB[i] = (unsigned long long)(uintptr_t)(BT + (size_t)(n0 + r) * Kp  + kc);
    ldsA[i] = asBase + (unsigned)(r * 64 + kc) * 2;
    ldsB[i] = bsBase + (unsigned)(r * 64 + kc) * 2;
  }

  auto stage = [&](unsigned bufOff) {
    #pragma unroll
    for (int i = 0; i < 4; ++i) ASYNC_B128(ldsA[i] + bufOff, gaA[i]);
    #pragma unroll
    for (int i = 0; i < 4; ++i) ASYNC_B128(ldsB[i] + bufOff, gaB[i]);
    #pragma unroll
    for (int i = 0; i < 4; ++i) { gaA[i] += 128; gaB[i] += 128; }
  };

  auto compute = [&](int buf) {
    const unsigned base = (unsigned)buf * (128 * 64);   // elements
    #pragma unroll
    for (int kh = 0; kh < 2; ++kh) {
      const v8bf* ap = (const v8bf*)(&As[0][0] + base +
                                     (wave * 16 + am) * 64 + kh * 32);
      v16bf a = SHUF16(ap[cl], ap[cl + 2]);
      v16bf bfr[8];
      #pragma unroll
      for (int j = 0; j < 8; ++j) {
        const v8bf* bp = (const v8bf*)(&Bs[0][0] + base +
                                       (j * 16 + am) * 64 + kh * 32);
        bfr[j] = SHUF16(bp[cl], bp[cl + 2]);
      }
      #pragma unroll
      for (int j = 0; j < 8; ++j)
        acc[j] = __builtin_amdgcn_wmma_f32_16x16x32_bf16(
            false, a, false, bfr[j], (short)0, acc[j], false, false);
    }
  };

  const int nk = Kp >> 6;               // Kp multiple of 64
  stage(0u);                            // prologue: stage 0 -> buffer 0

  // steady state: branch-free (issue next, overlap wait, compute current)
  for (int it = 0; it < nk - 1; ++it) {
    stage((it & 1) ? 0u : 16384u);      // next stage -> other buffer (bytes)
    asm volatile("s_wait_asynccnt 0x8" ::: "memory");   // stage 'it' complete
    __syncthreads();
    compute(it & 1);
    __syncthreads();
  }
  asm volatile("s_wait_asynccnt 0x0" ::: "memory");     // epilogue
  __syncthreads();
  compute((nk - 1) & 1);

  // store: VGPR r -> row r + 8*(lane>=16), col = lane&15 within 16-col group
  const int mo = m0 + wave * 16 + cl * 8;
  #pragma unroll
  for (int j = 0; j < 8; ++j) {
    const int col = n0 + j * 16 + am;   // always < Nc (Nc multiple of 128)
    #pragma unroll
    for (int r = 0; r < 8; ++r) {
      int m = mo + r;
      if (m < M) C[(size_t)m * Nc + col] = acc[j][r];
    }
  }
}

// ---------------------------------------------------------------------------
__global__ void node_attn_proj(const float* __restrict__ feat,
                               const float* __restrict__ al,
                               const float* __restrict__ ar,
                               float* __restrict__ el, float* __restrict__ er) {
  int i = blockIdx.x * blockDim.x + threadIdx.x;   // n*HEADS + h
  if (i >= N_NODES * HEADS) return;
  int h = i & 7;
  const float* f = feat + (size_t)i * HID;
  const float* a = al + h * HID;
  const float* b = ar + h * HID;
  float sl = 0.f, sr = 0.f;
  #pragma unroll 8
  for (int d = 0; d < HID; ++d) { float v = f[d]; sl += v * a[d]; sr += v * b[d]; }
  el[i] = sl; er[i] = sr;
}

__global__ void fill_f32(float* p, float v, int n) {
  int i = blockIdx.x * blockDim.x + threadIdx.x;
  if (i < n) p[i] = v;
}

__global__ void edge_logits(const int* __restrict__ src, const int* __restrict__ dst,
                            const float* __restrict__ el, const float* __restrict__ er,
                            float* __restrict__ ebuf, float* __restrict__ mx) {
  int i = blockIdx.x * blockDim.x + threadIdx.x;
  if (i >= N_EDGES * HEADS) return;
  int e = i >> 3, h = i & 7;
  int s = src[e], d = dst[e];
  float x = el[s * 8 + h] + er[d * 8 + h];
  x = x > 0.f ? x : 0.2f * x;
  ebuf[i] = x;
  atomicMaxF(&mx[d * 8 + h], x);
}

__global__ void finitize(float* m, int n) {
  int i = blockIdx.x * blockDim.x + threadIdx.x;
  if (i >= n) return;
  float v = m[i];
  if (!(v > -3.0e38f && v < 3.0e38f)) m[i] = 0.f;
}

__global__ void edge_exp(const int* __restrict__ dst, float* __restrict__ ebuf,
                         const float* __restrict__ mx, float* __restrict__ den) {
  int i = blockIdx.x * blockDim.x + threadIdx.x;
  if (i >= N_EDGES * HEADS) return;
  int e = i >> 3, h = i & 7;
  int d = dst[e];
  float ex = __expf(ebuf[i] - mx[d * 8 + h]);
  ebuf[i] = ex;
  atomicAdd(&den[d * 8 + h], ex);
}

__global__ __launch_bounds__(256)
void edge_aggregate(const int* __restrict__ src, const int* __restrict__ dst,
                    const float* __restrict__ ebuf, const float* __restrict__ den,
                    const float* __restrict__ feat, float* __restrict__ outz) {
  int e = blockIdx.x;
  int s = src[e], d = dst[e];
  for (int dd = threadIdx.x; dd < D1; dd += 256) {
    int h = dd >> 6;
    float alpha = ebuf[e * 8 + h] / fmaxf(den[d * 8 + h], 1e-9f);
    atomicAdd(&outz[(size_t)d * D1 + dd], alpha * feat[(size_t)s * D1 + dd]);
  }
}

__global__ void bias_elu(float* __restrict__ z, const float* __restrict__ b) {
  int i = blockIdx.x * blockDim.x + threadIdx.x;
  if (i >= N_NODES * D1) return;
  float x = z[i] + b[i & (D1 - 1)];
  z[i] = x > 0.f ? x : (__expf(x) - 1.f);
}

__global__ __launch_bounds__(128)
void sem_proj(const float* __restrict__ z, const float* __restrict__ W1,
              const float* __restrict__ b1, const float* __restrict__ W2,
              float* __restrict__ wsum) {
  __shared__ float red[SA_HIDN];
  int n = blockIdx.x;
  int j = threadIdx.x;
  const float* zr = z + (size_t)n * D1;
  float acc = b1[j];
  for (int i = 0; i < D1; ++i) acc += zr[i] * W1[i * SA_HIDN + j];
  red[j] = tanhf(acc) * W2[j];
  __syncthreads();
  for (int s = SA_HIDN / 2; s > 0; s >>= 1) {
    if (j < s) red[j] += red[j + s];
    __syncthreads();
  }
  if (j == 0) atomicAdd(wsum, red[0]);
}

__global__ void sem_beta(float* ws) {
  float w0 = ws[0] / (float)N_NODES, w1 = ws[1] / (float)N_NODES;
  float m = fmaxf(w0, w1);
  float e0 = __expf(w0 - m), e1 = __expf(w1 - m);
  float s = e0 + e1;
  ws[2] = e0 / s; ws[3] = e1 / s;
}

__global__ void sem_combine(const float* __restrict__ z, const float* __restrict__ ws,
                            float* __restrict__ x) {
  int i = blockIdx.x * blockDim.x + threadIdx.x;
  if (i >= N_NODES * D1) return;
  x[i] = ws[2] * z[i] + ws[3] * z[(size_t)N_NODES * D1 + i];
}

__global__ void predict_sigmoid(const float* __restrict__ x, const float* __restrict__ pW,
                                const float* __restrict__ pb, float* __restrict__ out) {
  int i = blockIdx.x * blockDim.x + threadIdx.x;
  if (i >= N_NODES * OUT_DIM) return;
  int n = i / OUT_DIM, o = i % OUT_DIM;
  const float* xr = x + (size_t)n * D1;
  float acc = pb[o];
  for (int k = 0; k < D1; ++k) acc += xr[k] * pW[k * OUT_DIM + o];
  out[i] = 1.f / (1.f + __expf(-acc));
}

// ---------------------------------------------------------------------------
extern "C" void kernel_launch(void* const* d_in, const int* in_sizes, int n_in,
                              void* d_out, int out_size, void* d_ws, size_t ws_size,
                              hipStream_t stream) {
  const float* adj    = (const float*)d_in[0];
  const float* h      = (const float*)d_in[1];
  const int*   src[2] = { (const int*)d_in[2], (const int*)d_in[4] };
  const int*   dst[2] = { (const int*)d_in[3], (const int*)d_in[5] };
  const float* lW[2]  = { (const float*)d_in[6],  (const float*)d_in[13] };
  const float* lal[2] = { (const float*)d_in[7],  (const float*)d_in[14] };
  const float* lar[2] = { (const float*)d_in[8],  (const float*)d_in[15] };
  const float* lb[2]  = { (const float*)d_in[9],  (const float*)d_in[16] };
  const float* saW1[2]= { (const float*)d_in[10], (const float*)d_in[17] };
  const float* sab1[2]= { (const float*)d_in[11], (const float*)d_in[18] };
  const float* saW2[2]= { (const float*)d_in[12], (const float*)d_in[19] };
  const float* pW = (const float*)d_in[20];
  const float* pb = (const float*)d_in[21];
  float* out = (float*)d_out;

  // ---- workspace layout ----
  float* ws = (float*)d_ws;
  const size_t ND = (size_t)N_NODES * D1;
  float* X    = ws;                          // N*512 f32
  float* FEAT = X    + ND;
  float* Z    = FEAT + ND;                   // 2 * N*512
  float* EL   = Z    + 2 * ND;
  float* ER   = EL   + (size_t)N_NODES * 8;
  float* MX   = ER   + (size_t)N_NODES * 8;
  float* DEN  = MX   + (size_t)N_NODES * 8;
  float* EBUF = DEN  + (size_t)N_NODES * 8;  // E*8
  float* WS   = EBUF + (size_t)N_EDGES * 8;  // 4 scalars
  // bf16 region (16B aligned: all offsets are multiples of 8 floats)
  unsigned short* ADJB = (unsigned short*)(WS + 8);        // [MP][KP_ADJ]
  unsigned short* HTB  = ADJB + (size_t)MP * KP_ADJ;       // [384][KP_ADJ]
  unsigned short* XB   = HTB  + (size_t)DIM_IN * KP_ADJ;   // [MP][<=512]
  unsigned short* WTB  = XB   + (size_t)MP * D1;           // [512][<=512]

  const dim3 B256(256);
  const int gNH = (N_NODES * HEADS + 255) / 256;
  const int gEH = (N_EDGES * HEADS + 255) / 256;
  const int gND = (int)((ND + 255) / 256);
  const int gMy = MP / 128;   // 79

  // x = adj @ h : convert adj (padded) and h^T (padded K), then WMMA GEMM
  {
    size_t tA = (size_t)MP * KP_ADJ;
    conv_pad_bf16<<<(int)((tA + 255) / 256), B256, 0, stream>>>(
        adj, ADJB, N_NODES, N_NODES, MP, KP_ADJ);
    size_t tH = (size_t)DIM_IN * KP_ADJ;
    conv_padT_bf16<<<(int)((tH + 255) / 256), B256, 0, stream>>>(
        h, HTB, N_NODES, DIM_IN, KP_ADJ);
    gemm_bf16_async<<<dim3(DIM_IN / 128, gMy), B256, 0, stream>>>(
        ADJB, HTB, X, N_NODES, DIM_IN, KP_ADJ, KP_ADJ);
  }

  for (int layer = 0; layer < 2; ++layer) {
    const int din = (layer == 0) ? DIM_IN : D1;
    size_t tX = (size_t)MP * din;
    conv_pad_bf16<<<(int)((tX + 255) / 256), B256, 0, stream>>>(
        X, XB, N_NODES, din, MP, din);
    for (int p = 0; p < 2; ++p) {
      float* Zp = Z + (size_t)p * ND;
      size_t tW = (size_t)D1 * din;
      conv_padT_bf16<<<(int)((tW + 255) / 256), B256, 0, stream>>>(
          lW[layer] + (size_t)p * din * D1, WTB, din, D1, din);
      gemm_bf16_async<<<dim3(D1 / 128, gMy), B256, 0, stream>>>(
          XB, WTB, FEAT, N_NODES, D1, din, din);
      node_attn_proj<<<gNH, B256, 0, stream>>>(
          FEAT, lal[layer] + p * HEADS * HID, lar[layer] + p * HEADS * HID, EL, ER);
      fill_f32<<<gNH, B256, 0, stream>>>(MX, -INFINITY, N_NODES * HEADS);
      fill_f32<<<gNH, B256, 0, stream>>>(DEN, 0.f, N_NODES * HEADS);
      edge_logits<<<gEH, B256, 0, stream>>>(src[p], dst[p], EL, ER, EBUF, MX);
      finitize<<<gNH, B256, 0, stream>>>(MX, N_NODES * HEADS);
      edge_exp<<<gEH, B256, 0, stream>>>(dst[p], EBUF, MX, DEN);
      fill_f32<<<gND, B256, 0, stream>>>(Zp, 0.f, (int)ND);
      edge_aggregate<<<N_EDGES, B256, 0, stream>>>(src[p], dst[p], EBUF, DEN, FEAT, Zp);
      bias_elu<<<gND, B256, 0, stream>>>(Zp, lb[layer] + p * D1);
    }
    fill_f32<<<1, 32, 0, stream>>>(WS, 0.f, 4);
    sem_proj<<<N_NODES, SA_HIDN, 0, stream>>>(Z,      saW1[layer], sab1[layer], saW2[layer], WS + 0);
    sem_proj<<<N_NODES, SA_HIDN, 0, stream>>>(Z + ND, saW1[layer], sab1[layer], saW2[layer], WS + 1);
    sem_beta<<<1, 1, 0, stream>>>(WS);
    sem_combine<<<gND, B256, 0, stream>>>(Z, WS, X);
  }

  predict_sigmoid<<<(N_NODES * OUT_DIM + 255) / 256, B256, 0, stream>>>(X, pW, pb, out);
}
